// HardLSTM_66159676227820
// MI455X (gfx1250) — matrix operand order repaired and checked by reading, not compiled, not run on hardware
//
#include <hip/hip_runtime.h>

typedef __attribute__((ext_vector_type(16))) __bf16 v16bf;
typedef __attribute__((ext_vector_type(8)))  __bf16 v8bf;
typedef __attribute__((ext_vector_type(8)))  float  v8f;

union V16U { v16bf v; v8bf h[2]; };

#define WMMA_BF16(A_, B_, C_) \
  __builtin_amdgcn_wmma_f32_16x16x32_bf16(false, (A_), false, (B_), (short)0, (C_), false, false)

static constexpr int T_  = 512;
static constexpr int Bn_ = 32;
static constexpr int H_  = 1024;
static constexpr int G4_ = 4096;     // 4*H
static constexpr int NB_ = 64;       // persistent blocks per direction (64*16 = H columns)
static constexpr int WROW_ = 1032;   // 1024 + 8 bf16 pad (kills LDS bank conflicts)
static constexpr int SMEM_BYTES = 64 * WROW_ * 2 /*w_hh slice*/ + 32 * 68 * 4 /*gates*/ + 32 * 16 * 4 /*c*/;

// ---------------------------------------------------------------- convert
__global__ __launch_bounds__(256) void cvt_f32_to_bf16(const float* __restrict__ in,
                                                       __bf16* __restrict__ out, long n) {
  long i = (long)blockIdx.x * blockDim.x + threadIdx.x;
  long stride = (long)gridDim.x * blockDim.x;
  for (; i < n; i += stride) out[i] = (__bf16)in[i];
}

// ---------------------------------------------------------------- GEMM: C[M,4096] = A[M,K] * W[4096,K]^T + bias
// grid: (N/64, M/32, 2 dirs), block 256 (8 waves: 2 Mtiles x 4 Ntiles)
__global__ __launch_bounds__(256) void gemm_bias_wmma(
    const __bf16* __restrict__ A, const __bf16* __restrict__ W,
    const float* __restrict__ bias, float* __restrict__ C,
    int K, long wDirStride, long bDirStride, long cDirStride) {
  const int dir = blockIdx.z;
  const __bf16* Wd = W + (long)dir * wDirStride;
  const float*  bd = bias + (long)dir * bDirStride;
  float*        Cd = C + (long)dir * cDirStride;

  const int lane = threadIdx.x & 31;
  const int wave = threadIdx.x >> 5;
  const int mt = wave >> 2;                 // 0..1
  const int nt = wave & 3;                  // 0..3
  const int m0 = blockIdx.y * 32 + mt * 16;
  const int n0 = blockIdx.x * 64 + nt * 16;
  const int lrow = lane & 15;
  const int lhi  = lane >> 4;

  const __bf16* aRow = A  + (long)(m0 + lrow) * K;
  const __bf16* wRow = Wd + (long)(n0 + lrow) * K;

  v8f acc = {};
  for (int k0 = 0; k0 < K; k0 += 32) {
    V16U a, b;
    a.h[0] = *(const v8bf*)(aRow + k0 + lhi * 8);        // K {0..7} / {8..15}
    a.h[1] = *(const v8bf*)(aRow + k0 + lhi * 8 + 16);   // K {16..23} / {24..31}
    b.h[0] = *(const v8bf*)(wRow + k0 + lhi * 8);        // K {0..7} / {8..15}
    b.h[1] = *(const v8bf*)(wRow + k0 + lhi * 8 + 16);   // K {16..23} / {24..31}
    acc = WMMA_BF16(a.v, b.v, acc);
  }
  const float bb = bd[n0 + lrow];
  float* cBase = Cd + (long)(m0 + lhi * 8) * G4_ + (n0 + lrow);
#pragma unroll
  for (int r = 0; r < 8; ++r) cBase[(long)r * G4_] = acc[r] + bb;
}

// ---------------------------------------------------------------- persistent scan
__device__ __forceinline__ void dir_barrier(unsigned* arrive, unsigned* phase, unsigned expect) {
  __syncthreads();
  if (threadIdx.x == 0) {
    __threadfence();                                   // release h/c writes
    unsigned t = atomicAdd(arrive, 1u);
    if (t == NB_ - 1) {
      __hip_atomic_store(arrive, 0u, __ATOMIC_RELAXED, __HIP_MEMORY_SCOPE_AGENT);
      __threadfence();
      atomicAdd(phase, 1u);
    } else {
      while (__hip_atomic_load(phase, __ATOMIC_RELAXED, __HIP_MEMORY_SCOPE_AGENT) < expect)
        __builtin_amdgcn_s_sleep(1);
    }
    __threadfence();                                   // acquire: invalidate
  }
  __syncthreads();
}

__device__ __forceinline__ float hsig(float x) { return fminf(fmaxf(0.2f * x + 0.5f, 0.f), 1.f); }
__device__ __forceinline__ float htanh(float x) { return fminf(fmaxf(x, -1.f), 1.f); }

// grid = 2*NB_ (dir-major), block 256 (8 waves: 2 Mtiles x 4 gates)
__global__ __launch_bounds__(256) void hardlstm_scan(
    const float* __restrict__ gx, long gxDirStride,
    const __bf16* __restrict__ whh, long whhDirStride,
    const float* __restrict__ h0, const float* __restrict__ c0, int stateOff,
    __bf16* __restrict__ hbuf,                    // [2 dirs][2 phase][Bn_*H_]
    __bf16* __restrict__ ybf,                     // layer0 output (bf16) or null
    float* __restrict__ yf32,                     // layer1 output (f32, d_out) or null
    float* __restrict__ hn, float* __restrict__ cn, int hnOff,
    unsigned* __restrict__ bar) {
  extern __shared__ char smem[];
  __bf16* wlds = (__bf16*)smem;                           // [64][WROW_]
  float*  gates = (float*)(smem + 64 * WROW_ * 2);        // [32][68]
  float*  cbuf  = (float*)((char*)gates + 32 * 68 * 4);   // [32][16]

  const int dir = blockIdx.x / NB_;
  const int blk = blockIdx.x % NB_;
  const int j0  = blk * 16;
  const int tid = threadIdx.x;
  const int lane = tid & 31, wave = tid >> 5;
  const int mt = wave & 1, gg = wave >> 1;                // gate 0..3
  const int lrow = lane & 15, lhi = lane >> 4;

  const float*  gxD  = gx  + (long)dir * gxDirStride;
  const __bf16* whhD = whh + (long)dir * whhDirStride;
  __bf16* hA = hbuf + (long)dir * (2 * Bn_ * H_);
  __bf16* hB = hA + Bn_ * H_;
  unsigned* arrive = bar + dir * 2;
  unsigned* phase  = arrive + 1;

  // stage this block's w_hh slice (4 gates x 16 cols x 1024 K, bf16) into LDS
  for (int idx = tid; idx < 64 * (H_ / 8); idx += 256) {
    int row = idx / (H_ / 8);
    int kc  = (idx % (H_ / 8)) * 8;
    int g = row >> 4, r = row & 15;
    *(v8bf*)(wlds + row * WROW_ + kc) =
        *(const v8bf*)(whhD + (long)(g * H_ + j0 + r) * H_ + kc);
  }
  // init h (phase-0 buffer, own slice) and resident c slice
  for (int e = tid; e < Bn_ * 16; e += 256) {
    int b = e >> 4, j = e & 15;
    long si = (long)stateOff + (long)dir * Bn_ * H_ + (long)b * H_ + j0 + j;
    hA[b * H_ + j0 + j] = (__bf16)h0[si];
    cbuf[b * 16 + j] = c0[si];
  }
  unsigned expect = 0;
  dir_barrier(arrive, phase, ++expect);

  const __bf16* bRowL = wlds + (gg * 16 + lrow) * WROW_;
  for (int s = 0; s < T_; ++s) {
    const int tt = (dir == 0) ? s : (T_ - 1 - s);
    const __bf16* hRead  = (s & 1) ? hB : hA;
    __bf16*       hWrite = (s & 1) ? hA : hB;

    // ---- h @ w_hh^T slice via WMMA (M=32, N=64, K=1024)
    v8f acc = {};
    const __bf16* aRow = hRead + (long)(mt * 16 + lrow) * H_;
    for (int k0 = 0; k0 < H_; k0 += 32) {
      V16U a, b;
      a.h[0] = *(const v8bf*)(aRow + k0 + lhi * 8);
      a.h[1] = *(const v8bf*)(aRow + k0 + lhi * 8 + 16);
      b.h[0] = *(const v8bf*)(bRowL + k0 + lhi * 8);
      b.h[1] = *(const v8bf*)(bRowL + k0 + lhi * 8 + 16);
      acc = WMMA_BF16(a.v, b.v, acc);
    }
    {
      int col = gg * 16 + lrow;
#pragma unroll
      for (int r = 0; r < 8; ++r) gates[(mt * 16 + r + lhi * 8) * 68 + col] = acc[r];
    }
    __syncthreads();

    // ---- fused gates + cell update for own 32x16 slice
    for (int e = tid; e < Bn_ * 16; e += 256) {
      int b = e >> 4, j = e & 15;
      const float* gxr = gxD + ((long)tt * Bn_ + b) * G4_ + (j0 + j);
      float iv = hsig (gxr[0]        + gates[b * 68 + j]);
      float fv = hsig (gxr[H_]       + gates[b * 68 + 16 + j]);
      float gv = htanh(gxr[2 * H_]   + gates[b * 68 + 32 + j]);
      float ov = hsig (gxr[3 * H_]   + gates[b * 68 + 48 + j]);
      float cy = fv * cbuf[b * 16 + j] + iv * gv;
      cbuf[b * 16 + j] = cy;
      float hy = ov * htanh(cy);
      hWrite[b * H_ + j0 + j] = (__bf16)hy;
      long yi = ((long)tt * Bn_ + b) * (2 * H_) + (long)dir * H_ + j0 + j;
      if (ybf)  ybf[yi]  = (__bf16)hy;
      if (yf32) yf32[yi] = hy;
      if (s == T_ - 1) {
        long oi = (long)(hnOff + dir) * Bn_ * H_ + (long)b * H_ + j0 + j;
        hn[oi] = hy;
        cn[oi] = cy;
      }
    }
    dir_barrier(arrive, phase, ++expect);
  }
}

// ---------------------------------------------------------------- host
extern "C" void kernel_launch(void* const* d_in, const int* in_sizes, int n_in,
                              void* d_out, int out_size, void* d_ws, size_t ws_size,
                              hipStream_t stream) {
  const float* x     = (const float*)d_in[0];
  const float* h0    = (const float*)d_in[1];
  const float* c0    = (const float*)d_in[2];
  const float* w_ih0 = (const float*)d_in[3];
  const float* w_hh0 = (const float*)d_in[4];
  const float* b0    = (const float*)d_in[5];
  const float* w_ih1 = (const float*)d_in[6];
  const float* w_hh1 = (const float*)d_in[7];
  const float* b1    = (const float*)d_in[8];

  const long TB = (long)T_ * Bn_;                  // 16384
  // workspace layout (bytes, 256-aligned)
  size_t off = 0;
  auto take = [&](size_t bytes) { size_t p = off; off += (bytes + 255) & ~(size_t)255; return p; };
  char* ws = (char*)d_ws;
  float*  gx   = (float*)(ws + take(2 * TB * (size_t)G4_ * 4));       // [2][T*B][4H]
  __bf16* xb   = (__bf16*)(ws + take(TB * 512 * 2));                  // x bf16
  __bf16* y0b  = (__bf16*)(ws + take(TB * 2048 * 2));                 // layer0 output bf16
  __bf16* wihb = (__bf16*)(ws + take(2ul * G4_ * 2048 * 2));          // max(I,2H)=2048
  __bf16* whhb = (__bf16*)(ws + take(2ul * G4_ * H_ * 2));
  __bf16* hbuf = (__bf16*)(ws + take(2ul * 2 * Bn_ * H_ * 2));
  unsigned* bar = (unsigned*)(ws + take(256));

  float* y1 = (float*)d_out;                                          // [T][B][2H]
  float* hn = y1 + (long)T_ * Bn_ * 2 * H_;                           // [4][B][H]
  float* cn = hn + 4l * Bn_ * H_;

  hipFuncSetAttribute((const void*)hardlstm_scan,
                      hipFuncAttributeMaxDynamicSharedMemorySize, SMEM_BYTES);

  dim3 blk(256);
  // ---- layer 0
  cvt_f32_to_bf16<<<1024, blk, 0, stream>>>(x, xb, TB * 512);
  cvt_f32_to_bf16<<<1024, blk, 0, stream>>>(w_ih0, wihb, 2l * G4_ * 512);
  cvt_f32_to_bf16<<<1024, blk, 0, stream>>>(w_hh0, whhb, 2l * G4_ * H_);
  gemm_bias_wmma<<<dim3(G4_ / 64, TB / 32, 2), blk, 0, stream>>>(
      xb, wihb, b0, gx, 512, (long)G4_ * 512, G4_, TB * (long)G4_);
  hipMemsetAsync(bar, 0, 256, stream);
  hardlstm_scan<<<2 * NB_, blk, SMEM_BYTES, stream>>>(
      gx, TB * (long)G4_, whhb, (long)G4_ * H_,
      h0, c0, /*stateOff=*/0, hbuf,
      /*ybf=*/y0b, /*yf32=*/nullptr, hn, cn, /*hnOff=*/0, bar);

  // ---- layer 1
  cvt_f32_to_bf16<<<1024, blk, 0, stream>>>(w_ih1, wihb, 2l * G4_ * 2048);
  cvt_f32_to_bf16<<<1024, blk, 0, stream>>>(w_hh1, whhb, 2l * G4_ * H_);
  gemm_bias_wmma<<<dim3(G4_ / 64, TB / 32, 2), blk, 0, stream>>>(
      y0b, wihb, b1, gx, 2048, (long)G4_ * 2048, G4_, TB * (long)G4_);
  hipMemsetAsync(bar, 0, 256, stream);
  hardlstm_scan<<<2 * NB_, blk, SMEM_BYTES, stream>>>(
      gx, TB * (long)G4_, whhb, (long)G4_ * H_,
      h0, c0, /*stateOff=*/2 * Bn_ * H_, hbuf,
      /*ybf=*/nullptr, /*yf32=*/y1, hn, cn, /*hnOff=*/2, bar);
}
